// AdvancedGCN_80625126080954
// MI455X (gfx1250) — matrix operand order repaired and checked by reading, not compiled, not run on hardware
//
#include <hip/hip_runtime.h>
#include <math.h>

typedef __attribute__((ext_vector_type(16))) _Float16 v16h;
typedef __attribute__((ext_vector_type(8)))  float    v8f;

// ---------------- degree / misc elementwise kernels ----------------
__global__ void k_init_deg(float* deg, int n) {
  int i = blockIdx.x * blockDim.x + threadIdx.x;
  if (i < n) deg[i] = 1.0f;                       // self loop
}
__global__ void k_count_deg(const int* __restrict__ ei, float* deg, int E) {
  int e = blockIdx.x * blockDim.x + threadIdx.x;
  if (e < E) atomicAdd(&deg[ei[E + e]], 1.0f);    // dst = ei[1][e]
}
__global__ void k_rsqrt_deg(float* deg, int n) {
  int i = blockIdx.x * blockDim.x + threadIdx.x;
  if (i < n) deg[i] = rsqrtf(deg[i]);
}
__global__ void k_zero(float* p, int n) {
  int i = blockIdx.x * blockDim.x + threadIdx.x;
  if (i < n) p[i] = 0.0f;
}

// ---------------- weight pre-pack: W[K x M] f32 -> WMMA B-operand f16 layout ----------------
// Packed index: (((mt*KC + kc)*32 + lane)*16 + i)  maps to  W[(kc*32 + laneHi*8 + i + (i>=8 ? 8 : 0))*M + mt*16 + laneLo]
// so each lane's 16-half B fragment is one contiguous 32B chunk (2x global_load_b128).
template <int K, int M>
__global__ void k_pack_w(const float* __restrict__ W, _Float16* __restrict__ Wp) {
  constexpr int KC = K / 32;
  int idx = blockIdx.x * blockDim.x + threadIdx.x;
  if (idx >= K * M) return;
  int i    = idx & 15;
  int lane = (idx >> 4) & 31;
  int rest = idx >> 9;
  int kc   = rest % KC;
  int mt   = rest / KC;
  int laneLo = lane & 15;
  int laneHi = lane >> 4;
  int k = kc * 32 + laneHi * 8 + i + (i >= 8 ? 8 : 0);
  int n = mt * 16 + laneLo;
  Wp[idx] = (_Float16)W[(size_t)k * M + n];
}

// ---------------- WMMA GEMM: Y[N x M] = X[N x K] @ W[K x M] (+bias) ----------------
// One wave computes a 16-row tile. A (16xK) held in VGPRs across all M-tiles.
// B comes pre-packed in f16 -> one v16h load per (m-tile, k-chunk) per lane.
template <int K, int M>
__global__ void k_gemm_wmma(const float* __restrict__ X, const _Float16* __restrict__ Wp,
                            const float* __restrict__ bias, float* __restrict__ Y,
                            int nrows) {
  const int lane = threadIdx.x & 31;
  const int wave = threadIdx.x >> 5;
  const int tile = blockIdx.x * (blockDim.x >> 5) + wave;
  const int row0 = tile * 16;
  if (row0 >= nrows) return;                     // wave-uniform: EXEC stays all-ones
  const int laneLo = lane & 15;
  const int laneHi = lane >> 4;
  const int r = row0 + laneLo;

  constexpr int KC = K / 32;
  v16h aReg[KC];
#pragma unroll
  for (int c = 0; c < KC; ++c) {
    // A 16x32 f16 layout: lane half selects K-halves {0-7,16-23} vs {8-15,24-31}
    const float* base = X + (size_t)r * K + c * 32 + laneHi * 8;
    float4 x0 = *(const float4*)(base);
    float4 x1 = *(const float4*)(base + 4);
    float4 x2 = *(const float4*)(base + 16);
    float4 x3 = *(const float4*)(base + 20);
    v16h a;
    a[0]  = (_Float16)x0.x; a[1]  = (_Float16)x0.y; a[2]  = (_Float16)x0.z; a[3]  = (_Float16)x0.w;
    a[4]  = (_Float16)x1.x; a[5]  = (_Float16)x1.y; a[6]  = (_Float16)x1.z; a[7]  = (_Float16)x1.w;
    a[8]  = (_Float16)x2.x; a[9]  = (_Float16)x2.y; a[10] = (_Float16)x2.z; a[11] = (_Float16)x2.w;
    a[12] = (_Float16)x3.x; a[13] = (_Float16)x3.y; a[14] = (_Float16)x3.z; a[15] = (_Float16)x3.w;
    aReg[c] = a;
  }

  const v16h* Bp = (const v16h*)Wp;
#pragma unroll
  for (int mt = 0; mt < M / 16; ++mt) {
    const int n = mt * 16 + laneLo;
    v8f acc = {};
#pragma unroll
    for (int c = 0; c < KC; ++c) {
      v16h b = Bp[(size_t)(mt * KC + c) * 32 + lane];   // contiguous 32B per lane
      acc = __builtin_amdgcn_wmma_f32_16x16x32_f16(false, aReg[c], false, b,
                                                   (short)0, acc, false, false);
    }
    const float bv = bias ? bias[n] : 0.0f;
    // C/D layout: VGPR v holds row (laneHi*8 + v), column = laneLo
#pragma unroll
    for (int v = 0; v < 8; ++v) {
      const int rr = row0 + laneHi * 8 + v;
      Y[(size_t)rr * M + n] = acc[v] + bv;
    }
  }
}

// ---------------- edge scatter: agg[dst] += dinv[src]*dinv[dst] * h[src] ----------------
// One thread per (edge, 4-feature chunk). dO in {128,64,32} -> dO/4 power of two.
__global__ void k_scatter(const int* __restrict__ ei, const float* __restrict__ h,
                          const float* __restrict__ dinv, float* __restrict__ agg,
                          int E, int sh /* log2(dO/4) */, int dO) {
  int idx = blockIdx.x * blockDim.x + threadIdx.x;
  int total = E << sh;
  if (idx >= total) return;
  int e  = idx >> sh;
  int f4 = (idx & ((1 << sh) - 1)) << 2;
  int src = ei[e];
  int dst = ei[E + e];
  float norm = dinv[src] * dinv[dst];
  float4 v = *(const float4*)(h + (size_t)src * dO + f4);
  float* out = agg + (size_t)dst * dO + f4;
  atomicAdd(out + 0, v.x * norm);
  atomicAdd(out + 1, v.y * norm);
  atomicAdd(out + 2, v.z * norm);
  atomicAdd(out + 3, v.w * norm);
}

// ---------------- fused epilogue: relu(bn(agg + lin*dinv^2 + b)) + res ----------------
__global__ void k_fuse(const float* __restrict__ agg, const float* __restrict__ lin,
                       const float* __restrict__ bias,
                       const float* __restrict__ g, const float* __restrict__ bt,
                       const float* __restrict__ bm, const float* __restrict__ bv,
                       const float* __restrict__ dinv, const float* __restrict__ res,
                       float* __restrict__ out, int Nn, int dO) {
  int idx = blockIdx.x * blockDim.x + threadIdx.x;
  if (idx >= Nn * dO) return;
  int n = idx / dO;
  int f = idx - n * dO;
  float di = dinv[n];
  float val = agg[idx] + lin[idx] * di * di + bias[f];
  val = (val - bm[f]) * rsqrtf(bv[f] + 1e-5f) * g[f] + bt[f];
  val = fmaxf(val, 0.0f);
  out[idx] = val + res[idx];
}

// ---------------- classifier: 32 -> 16 (relu) -> 2, log_softmax ----------------
__global__ void k_classifier(const float* __restrict__ h,
                             const float* __restrict__ W1, const float* __restrict__ b1,
                             const float* __restrict__ W2, const float* __restrict__ b2,
                             float* __restrict__ out, int Nn) {
  int n = blockIdx.x * blockDim.x + threadIdx.x;
  if (n >= Nn) return;
  float x[32];
#pragma unroll
  for (int k = 0; k < 32; ++k) x[k] = h[(size_t)n * 32 + k];
  float l0 = b2[0], l1 = b2[1];
#pragma unroll
  for (int j = 0; j < 16; ++j) {
    float s = b1[j];
#pragma unroll
    for (int k = 0; k < 32; ++k) s += x[k] * W1[k * 16 + j];
    s = fmaxf(s, 0.0f);
    l0 += s * W2[j * 2 + 0];
    l1 += s * W2[j * 2 + 1];
  }
  float mx  = fmaxf(l0, l1);
  float lse = mx + logf(expf(l0 - mx) + expf(l1 - mx));
  out[(size_t)n * 2 + 0] = l0 - lse;
  out[(size_t)n * 2 + 1] = l1 - lse;
}

// ---------------- host orchestration ----------------
static inline int cdiv(int a, int b) { return (a + b - 1) / b; }

extern "C" void kernel_launch(void* const* d_in, const int* in_sizes, int n_in,
                              void* d_out, int out_size, void* d_ws, size_t ws_size,
                              hipStream_t stream) {
  const float* x   = (const float*)d_in[0];
  const int*   ei  = (const int*)d_in[1];
  const float* W0  = (const float*)d_in[2];
  const float* b0  = (const float*)d_in[3];
  const float* g0  = (const float*)d_in[4];
  const float* be0 = (const float*)d_in[5];
  const float* m0  = (const float*)d_in[6];
  const float* v0  = (const float*)d_in[7];
  const float* W1  = (const float*)d_in[8];
  const float* b1  = (const float*)d_in[9];
  const float* g1  = (const float*)d_in[10];
  const float* be1 = (const float*)d_in[11];
  const float* m1  = (const float*)d_in[12];
  const float* v1  = (const float*)d_in[13];
  const float* s1W = (const float*)d_in[14];
  const float* s1b = (const float*)d_in[15];
  const float* W2  = (const float*)d_in[16];
  const float* b2  = (const float*)d_in[17];
  const float* g2  = (const float*)d_in[18];
  const float* be2 = (const float*)d_in[19];
  const float* m2  = (const float*)d_in[20];
  const float* v2  = (const float*)d_in[21];
  const float* s2W = (const float*)d_in[22];
  const float* s2b = (const float*)d_in[23];
  const float* c1W = (const float*)d_in[24];
  const float* c1b = (const float*)d_in[25];
  const float* c2W = (const float*)d_in[26];
  const float* c2b = (const float*)d_in[27];

  const int N = in_sizes[0] / 128;
  const int E = in_sizes[1] / 2;

  // workspace: dinv[N] | A[N*128] | B[N*128] | Cg[N*128] | R[N*64] | Wp[16384 f16]
  float* dinv = (float*)d_ws;
  float* A    = dinv + N;
  float* B    = A + (size_t)N * 128;
  float* Cg   = B + (size_t)N * 128;
  float* R    = Cg + (size_t)N * 128;
  // align packed-weight buffer to 64B for v16h (b128) loads
  size_t wpOff = ((size_t)(R + (size_t)N * 64 - (float*)d_ws) * 4 + 63) & ~(size_t)63;
  _Float16* Wp = (_Float16*)((char*)d_ws + wpOff);

  const int T = 256;
  const int gemmBlocks = cdiv(N / 16, 8);  // 8 waves/block, 16 rows/wave

  // degrees -> dinv
  k_init_deg<<<cdiv(N, T), T, 0, stream>>>(dinv, N);
  k_count_deg<<<cdiv(E, T), T, 0, stream>>>(ei, dinv, E);
  k_rsqrt_deg<<<cdiv(N, T), T, 0, stream>>>(dinv, N);

  // ---- layer 0: 128 -> 128, identity skip (res = x) ----
  k_pack_w<128, 128><<<cdiv(128 * 128, T), T, 0, stream>>>(W0, Wp);
  k_gemm_wmma<128, 128><<<gemmBlocks, T, 0, stream>>>(x, Wp, nullptr, B, N);
  k_zero<<<cdiv(N * 128, T), T, 0, stream>>>(Cg, N * 128);
  k_scatter<<<cdiv(E * 32, T), T, 0, stream>>>(ei, B, dinv, Cg, E, 5, 128);
  k_fuse<<<cdiv(N * 128, T), T, 0, stream>>>(Cg, B, b0, g0, be0, m0, v0, dinv, x, A, N, 128);

  // ---- layer 1: 128 -> 64, linear skip ----
  k_pack_w<128, 64><<<cdiv(128 * 64, T), T, 0, stream>>>(s1W, Wp);
  k_gemm_wmma<128, 64><<<gemmBlocks, T, 0, stream>>>(A, Wp, s1b, R, N);    // res
  k_pack_w<128, 64><<<cdiv(128 * 64, T), T, 0, stream>>>(W1, Wp);
  k_gemm_wmma<128, 64><<<gemmBlocks, T, 0, stream>>>(A, Wp, nullptr, B, N);
  k_zero<<<cdiv(N * 64, T), T, 0, stream>>>(Cg, N * 64);
  k_scatter<<<cdiv(E * 16, T), T, 0, stream>>>(ei, B, dinv, Cg, E, 4, 64);
  k_fuse<<<cdiv(N * 64, T), T, 0, stream>>>(Cg, B, b1, g1, be1, m1, v1, dinv, R, A, N, 64);

  // ---- layer 2: 64 -> 32, linear skip ----
  k_pack_w<64, 32><<<cdiv(64 * 32, T), T, 0, stream>>>(s2W, Wp);
  k_gemm_wmma<64, 32><<<gemmBlocks, T, 0, stream>>>(A, Wp, s2b, R, N);     // res
  k_pack_w<64, 32><<<cdiv(64 * 32, T), T, 0, stream>>>(W2, Wp);
  k_gemm_wmma<64, 32><<<gemmBlocks, T, 0, stream>>>(A, Wp, nullptr, B, N);
  k_zero<<<cdiv(N * 32, T), T, 0, stream>>>(Cg, N * 32);
  k_scatter<<<cdiv(E * 8, T), T, 0, stream>>>(ei, B, dinv, Cg, E, 3, 32);
  k_fuse<<<cdiv(N * 32, T), T, 0, stream>>>(Cg, B, b2, g2, be2, m2, v2, dinv, R, Cg, N, 32);

  // ---- classifier + log_softmax ----
  k_classifier<<<cdiv(N, T), T, 0, stream>>>(Cg, c1W, c1b, c2W, c2b, (float*)d_out, N);
}
